// DCOMPNet_12884901888470
// MI455X (gfx1250) — matrix operand order repaired and checked by reading, not compiled, not run on hardware
//
#include <hip/hip_runtime.h>
#include <stdint.h>

#define B_    2
#define CIN_  32
#define C_    64
#define H_    352
#define W_    1216
#define HO_   176
#define WO_   608
#define NS_   10000
#define KNN_  7
#define HWO_  (HO_*WO_)        // 107008
#define NPIX_ (B_*HWO_)        // 214016
#define FIN_  131
#define HID_  65

typedef __attribute__((ext_vector_type(16))) __bf16 v16bf;
typedef __attribute__((ext_vector_type(8)))  float  v8f;
typedef unsigned int u32;
typedef __attribute__((ext_vector_type(4))) u32 v4u;
typedef __attribute__((ext_vector_type(8))) int  v8i;
typedef __attribute__((ext_vector_type(4))) int  v4i;

union Frag { uint4 u[2]; v16bf v; };

__device__ __forceinline__ __bf16 f2bf(float f) {
  u32 u = __builtin_bit_cast(u32, f);
  u32 r = (u + 0x7FFFu + ((u >> 16) & 1u)) >> 16;
  unsigned short s = (unsigned short)r;
  return __builtin_bit_cast(__bf16, s);
}

// ---- gfx1250 async global->LDS copy (ASYNCcnt path), with sync fallback ----
#if __has_builtin(__builtin_amdgcn_global_load_async_to_lds_b128)
#define HAS_ASYNC_LDS 1
typedef __attribute__((__vector_size__(16))) int v4si;
typedef __attribute__((address_space(1))) v4si* as1_v4si_p;  // global int4*
typedef __attribute__((address_space(3))) v4si* as3_v4si_p;  // LDS int4*
#else
#define HAS_ASYNC_LDS 0
#endif

__device__ __forceinline__ void async_copy_b128(const void* g, void* l) {
#if HAS_ASYNC_LDS
  __builtin_amdgcn_global_load_async_to_lds_b128((as1_v4si_p)g, (as3_v4si_p)l, 0, 0);
#else
  *(uint4*)l = *(const uint4*)g;
#endif
}

__device__ __forceinline__ void wait_async_lds() {
#if HAS_ASYNC_LDS
#if __has_builtin(__builtin_amdgcn_s_wait_asynccnt)
  __builtin_amdgcn_s_wait_asynccnt(0);
#else
  asm volatile("s_wait_asynccnt 0" ::: "memory");
#endif
#endif
}

// ---- Tensor Data Mover: pull a contiguous bf16 weight block into LDS ----
__device__ __forceinline__ void tdm_load_weights(const __bf16* gsrc, __bf16* lds, u32 nelem) {
#if __has_builtin(__builtin_amdgcn_tensor_load_to_lds)
  if ((threadIdx.x >> 5) == 0) {   // wave 0 issues the TDM op once
    u32 ldsOff = (u32)(uintptr_t)(void*)lds;
    uint64_t ga = (uint64_t)(uintptr_t)gsrc;
    // D# group0: count=1, lds_addr[63:32], global_addr[120:64], type=2 in [127:126]
    v4u g0 = { 1u, ldsOff, (u32)ga, ((u32)(ga >> 32) & 0x1FFFFFFu) | (2u << 30) };
    // D# group1: data_size=1 (2B); 1-D tensor: dim0 = tile_dim0 = nelem (< 65536)
    v8i g1 = { (int)(1u << 16),
               (int)((nelem & 0xFFFFu) << 16),          // tensor_dim0[15:0] << 16
               (int)((nelem >> 16) | (1u << 16)),       // tensor_dim0[31:16] | tensor_dim1=1
               (int)((nelem & 0xFFFFu) << 16),          // tile_dim0 << 16
               (int)1,                                   // tile_dim1=1, tile_dim2=0
               (int)nelem,                               // tensor_dim0_stride lo
               0, 0 };
    v4i gz = {0, 0, 0, 0};
#if defined(__clang_major__) && (__clang_major__ >= 23)
    v8i gz8 = {0, 0, 0, 0, 0, 0, 0, 0};
    __builtin_amdgcn_tensor_load_to_lds(g0, g1, gz, gz, gz8, 0);
#else
    __builtin_amdgcn_tensor_load_to_lds(g0, g1, gz, gz, 0);
#endif
    __builtin_amdgcn_s_wait_tensorcnt(0);
  }
  __syncthreads();
#else
  const uint4* s = (const uint4*)gsrc;
  uint4* d = (uint4*)lds;
  for (u32 i = threadIdx.x; i < nelem / 8u; i += blockDim.x) d[i] = s[i];
  __syncthreads();
#endif
}

// ---- weight packing: fp32 OIHW -> bf16 [kchunk][O][32] ----
__global__ void k_pack(const float* __restrict__ dc0w, const float* __restrict__ dc1w,
                       const float* __restrict__ rc0w, const float* __restrict__ rc1w,
                       const float* __restrict__ dc2w, const float* __restrict__ rc2w,
                       __bf16* __restrict__ wpk2, __bf16* __restrict__ wpk1) {
  int t = blockIdx.x * 256 + threadIdx.x;
  const int S2 = 4 * 9 * 64 * 32;   // 73728
  const int S1 = 2 * 18 * 64 * 32;  // 73728
  if (t < S2) {
    int cv = t / 18432, e = t % 18432;
    int tap = e / 2048, o = (e % 2048) / 32, ci = e % 32;
    const float* src = (cv == 0) ? dc0w : (cv == 1) ? dc1w : (cv == 2) ? rc0w : rc1w;
    wpk2[t] = f2bf(src[(o * CIN_ + ci) * 9 + tap]);
  } else if (t < S2 + S1) {
    int u = t - S2;
    int cv = u / 36864, e = u % 36864;
    int chunk = e / 2048, o = (e % 2048) / 32, ci2 = e % 32;
    int tap = chunk >> 1, ci = (chunk & 1) * 32 + ci2;
    const float* src = cv ? rc2w : dc2w;
    wpk1[u] = f2bf(src[(o * C_ + ci) * 9 + tap]);
  }
}

// ---- input conversion: NCHW fp32 -> NHWC bf16 ----
__global__ void k_cvt_in(const float* __restrict__ df, const float* __restrict__ rf,
                         __bf16* __restrict__ xd, __bf16* __restrict__ xr) {
  long t = (long)blockIdx.x * 256 + threadIdx.x;
  const long N1 = (long)B_ * H_ * W_ * CIN_;
  if (t >= 2 * N1) return;
  int br = (int)(t / N1);
  long u = t - (long)br * N1;
  int c = (int)(u % CIN_);
  long v = u / CIN_;
  int x = (int)(v % W_);
  long y2 = v / W_;
  int y = (int)(y2 % H_);
  int b = (int)(y2 / H_);
  const float* src = br ? rf : df;
  __bf16* dst = br ? xr : xd;
  dst[u] = f2bf(src[(((size_t)b * CIN_ + c) * H_ + y) * W_ + x]);
}

// ---- A-tile staging helpers (async double-buffer) ----
__device__ __forceinline__ void stage_s2(const __bf16* __restrict__ x, __bf16* dst,
                                         int pixBase, int kh, int kw) {
  for (int q = threadIdx.x; q < 128 * 4; q += 256) {
    int prow = q >> 2, seg = q & 3;
    int p = pixBase + prow;
    int bb = p / HWO_, off = p - bb * HWO_;
    int oh = off / WO_, ow = off - oh * WO_;
    int ih = oh * 2 - 1 + kh, iw = ow * 2 - 1 + kw;
    __bf16* l = dst + prow * 32 + seg * 8;
    if ((unsigned)ih < (unsigned)H_ && (unsigned)iw < (unsigned)W_) {
      async_copy_b128(x + (((size_t)bb * H_ + ih) * W_ + iw) * CIN_ + seg * 8, l);
    } else {
      uint4 z = {0u, 0u, 0u, 0u};
      *(uint4*)l = z;
    }
  }
}

__device__ __forceinline__ void stage_s1(const __bf16* __restrict__ x, __bf16* dst,
                                         int pixBase, int kh, int kw) {
  for (int q = threadIdx.x; q < 128 * 8; q += 256) {
    int prow = q >> 3, seg = q & 7;
    int p = pixBase + prow;
    int bb = p / HWO_, off = p - bb * HWO_;
    int oh = off / WO_, ow = off - oh * WO_;
    int ih = oh - 1 + kh, iw = ow - 1 + kw;
    __bf16* l = dst + prow * 64 + seg * 8;
    if ((unsigned)ih < (unsigned)HO_ && (unsigned)iw < (unsigned)WO_) {
      async_copy_b128(x + (((size_t)bb * HO_ + ih) * WO_ + iw) * C_ + seg * 8, l);
    } else {
      uint4 z = {0u, 0u, 0u, 0u};
      *(uint4*)l = z;
    }
  }
}

// ---- fused stride-2 convs (conv0 w/ relu + conv1) per branch, bf16 WMMA ----
__global__ __launch_bounds__(256, 1)
void k_conv_s2(const __bf16* __restrict__ xd, const __bf16* __restrict__ xr,
               const __bf16* __restrict__ wpk2,
               const float* __restrict__ dc0b, const float* __restrict__ dc1b,
               const float* __restrict__ rc0b, const float* __restrict__ rc1b,
               float* __restrict__ d0g, float* __restrict__ d1g,
               float* __restrict__ r0g, float* __restrict__ r1g) {
  const int br = blockIdx.y;
  const __bf16* x  = br ? xr : xd;
  const __bf16* ws = wpk2 + (size_t)br * 2 * 9 * 64 * 32;
  const float* b0  = br ? rc0b : dc0b;
  const float* b1  = br ? rc1b : dc1b;
  float* out0 = br ? r0g : d0g;
  float* out1 = br ? r1g : d1g;

  __shared__ __align__(16) __bf16 wlds[2 * 9 * 64 * 32];   // 72 KB
  __shared__ __align__(16) __bf16 atile[2][128 * 32];      // 2 x 8 KB

  tdm_load_weights(ws, wlds, 2 * 9 * 64 * 32);

  const int pixBase = blockIdx.x * 128;
  const int lane = threadIdx.x & 31;
  const int wave = threadIdx.x >> 5;
  const int mrow = wave * 16 + (lane & 15);
  const int ako  = (lane & 16) ? 16 : 0;  // A: K {0..15} vs {16..31}
  const int bko  = (lane & 16) ? 16 : 0;  // B: K {0..15} vs {16..31}

  v8f zero = {0.f, 0.f, 0.f, 0.f, 0.f, 0.f, 0.f, 0.f};
  v8f acc[2][4];
  for (int i = 0; i < 2; ++i)
    for (int j = 0; j < 4; ++j) acc[i][j] = zero;

  // prologue: stage tap 0 into buffer 0
  stage_s2(x, atile[0], pixBase, 0, 0);
  wait_async_lds();
  __syncthreads();

  for (int tap = 0; tap < 9; ++tap) {
    const int cur = tap & 1;
    if (tap + 1 < 9)  // prefetch next tap into the other buffer while computing
      stage_s2(x, atile[1 - cur], pixBase, (tap + 1) / 3, (tap + 1) % 3);

    const __bf16* arow = atile[cur] + mrow * 32;
    Frag af;
    af.u[0] = *(const uint4*)(arow + ako);
    af.u[1] = *(const uint4*)(arow + ako + 8);

    Frag bf[2][4];                     // batch all B-fragment LDS loads first
    for (int cw = 0; cw < 2; ++cw)
      for (int nt = 0; nt < 4; ++nt) {
        int n = nt * 16 + (lane & 15);
        const __bf16* wb = wlds + (((cw * 9 + tap) * 64 + n) * 32 + bko);
        bf[cw][nt].u[0] = ((const uint4*)wb)[0];
        bf[cw][nt].u[1] = ((const uint4*)wb)[1];
      }
    for (int cw = 0; cw < 2; ++cw)
      for (int nt = 0; nt < 4; ++nt)
        acc[cw][nt] = __builtin_amdgcn_wmma_f32_16x16x32_bf16(
            false, af.v, false, bf[cw][nt].v, (short)0, acc[cw][nt], false, false);

    wait_async_lds();
    __syncthreads();
  }

  for (int cw = 0; cw < 2; ++cw) {
    float* outp = cw ? out1 : out0;
    const float* bptr = cw ? b1 : b0;
    for (int nt = 0; nt < 4; ++nt) {
      int ch = nt * 16 + (lane & 15);
      float bi = bptr[ch];
      for (int j = 0; j < 8; ++j) {
        int m = j + ((lane & 16) ? 8 : 0);
        int p = pixBase + wave * 16 + m;
        int bb = p / HWO_, off = p - bb * HWO_;
        float v = acc[cw][nt][j] + bi;
        if (cw == 0) v = fmaxf(v, 0.0f);
        outp[((size_t)bb * C_ + ch) * HWO_ + off] = v;
      }
    }
  }
}

// ---- gather sampled features ----
__global__ void k_gather(const float* __restrict__ d0g, const float* __restrict__ r0g,
                         const int* __restrict__ sidxs,
                         float* __restrict__ dsg, float* __restrict__ rsg) {
  long t = (long)blockIdx.x * 256 + threadIdx.x;
  const long N1 = (long)B_ * C_ * NS_;
  if (t >= 2 * N1) return;
  int br = (int)(t / N1);
  long u = t - (long)br * N1;
  int n = (int)(u % NS_);
  int c = (int)((u / NS_) % C_);
  int b = (int)(u / ((long)NS_ * C_));
  int sid = sidxs[b * NS_ + n];
  const float* g = br ? r0g : d0g;
  float* o = br ? rsg : dsg;
  o[((size_t)b * C_ + c) * NS_ + n] = g[((size_t)b * C_ + c) * HWO_ + sid];
}

// ---- KNN attention: one wave per (b,n) sample ----
__global__ __launch_bounds__(256, 1)
void k_attn(const float* __restrict__ dsg, const float* __restrict__ rsg,
            const float* __restrict__ spoints, const int* __restrict__ nnidxs,
            const float* __restrict__ dw1, const float* __restrict__ db1,
            const float* __restrict__ dw2, const float* __restrict__ db2,
            const float* __restrict__ rw1, const float* __restrict__ rb1,
            const float* __restrict__ rw2, const float* __restrict__ rb2,
            const float* __restrict__ dbias, const float* __restrict__ rbias,
            float* __restrict__ dagg, float* __restrict__ ragg) {
  __shared__ float W1D[FIN_ * HID_];
  __shared__ float W1R[FIN_ * HID_];
  __shared__ float B1D[HID_], B1R[HID_], W2D[HID_], W2R[HID_];
  __shared__ float cen[8][FIN_ + 1];
  __shared__ float nb[8][FIN_ + 1];
  __shared__ float lgd[8][KNN_ + 1], lgr[8][KNN_ + 1];
  __shared__ int   nns[8][KNN_ + 1];

  for (int i = threadIdx.x; i < FIN_ * HID_; i += 256) { W1D[i] = dw1[i]; W1R[i] = rw1[i]; }
  for (int i = threadIdx.x; i < HID_; i += 256) {
    B1D[i] = db1[i]; B1R[i] = rb1[i]; W2D[i] = dw2[i]; W2R[i] = rw2[i];
  }
  __syncthreads();

  const int wave = threadIdx.x >> 5, lane = threadIdx.x & 31;
  const int s = blockIdx.x * 8 + wave;   // exact: grid = B*NS/8
  const int b = s / NS_, n = s % NS_;

  for (int t = 0; t < 2; ++t) {
    int c = lane + 32 * t;
    cen[wave][c]      = dsg[((size_t)b * C_ + c) * NS_ + n];
    cen[wave][64 + c] = rsg[((size_t)b * C_ + c) * NS_ + n];
  }
  if (lane < 3) cen[wave][128 + lane] = spoints[((size_t)b * 3 + lane) * NS_ + n];

  for (int k = 0; k < KNN_; ++k) {
    __syncthreads();
    int nn = nnidxs[((size_t)b * NS_ + n) * KNN_ + k];
    if (lane == 0) nns[wave][k] = nn;
    for (int t = 0; t < 2; ++t) {
      int c = lane + 32 * t;
      nb[wave][c]      = dsg[((size_t)b * C_ + c) * NS_ + nn] - cen[wave][c];
      nb[wave][64 + c] = rsg[((size_t)b * C_ + c) * NS_ + nn] - cen[wave][64 + c];
    }
    if (lane < 3)
      nb[wave][128 + lane] = spoints[((size_t)b * 3 + lane) * NS_ + nn] - cen[wave][128 + lane];
    __syncthreads();

    float pd = 0.f, pr = 0.f;
    for (int j = lane; j < HID_; j += 32) {
      float hd = B1D[j], hr = B1R[j];
      for (int f = 0; f < FIN_; ++f) {
        float xv = nb[wave][f];
        hd += xv * W1D[f * HID_ + j];
        hr += xv * W1R[f * HID_ + j];
      }
      hd = (hd > 0.f) ? hd : 0.2f * hd;
      hr = (hr > 0.f) ? hr : 0.2f * hr;
      pd += hd * W2D[j];
      pr += hr * W2R[j];
    }
    for (int o = 16; o > 0; o >>= 1) {
      pd += __shfl_xor(pd, o, 32);
      pr += __shfl_xor(pr, o, 32);
    }
    if (lane == 0) { lgd[wave][k] = pd + db2[0]; lgr[wave][k] = pr + rb2[0]; }
  }
  __syncthreads();

  float ad[KNN_], ar[KNN_];
  float mxd = -1e30f, mxr = -1e30f;
  for (int k = 0; k < KNN_; ++k) { mxd = fmaxf(mxd, lgd[wave][k]); mxr = fmaxf(mxr, lgr[wave][k]); }
  float sd = 0.f, sr = 0.f;
  for (int k = 0; k < KNN_; ++k) {
    ad[k] = __expf(lgd[wave][k] - mxd); sd += ad[k];
    ar[k] = __expf(lgr[wave][k] - mxr); sr += ar[k];
  }
  float isd = 1.f / sd, isr = 1.f / sr;

  for (int t = 0; t < 2; ++t) {
    int c = lane + 32 * t;
    float accd = 0.f, accr = 0.f;
    for (int k = 0; k < KNN_; ++k) {
      int nn = nns[wave][k];
      accd += ad[k] * isd * dsg[((size_t)b * C_ + c) * NS_ + nn];
      accr += ar[k] * isr * rsg[((size_t)b * C_ + c) * NS_ + nn];
    }
    dagg[((size_t)b * C_ + c) * NS_ + n] = accd + dbias[c];
    ragg[((size_t)b * C_ + c) * NS_ + n] = accr + rbias[c];
  }
}

// ---- d0 *= (1 - mask) ----
__global__ void k_mix(const int* __restrict__ masks, float* __restrict__ d0g, float* __restrict__ r0g) {
  long t = (long)blockIdx.x * 256 + threadIdx.x;
  const long N1 = (long)B_ * C_ * HWO_;
  if (t >= 2 * N1) return;
  int br = (int)(t / N1);
  long u = t - (long)br * N1;
  int pix = (int)(u % HWO_);
  int b = (int)(u / ((long)HWO_ * C_));
  float m = (float)masks[(size_t)b * HWO_ + pix];
  float* g = br ? r0g : d0g;
  g[u] *= (1.0f - m);
}

// ---- scatter aggregated features ----
__global__ void k_scatter(const int* __restrict__ sidxs,
                          const float* __restrict__ dagg, const float* __restrict__ ragg,
                          float* __restrict__ d0g, float* __restrict__ r0g) {
  long t = (long)blockIdx.x * 256 + threadIdx.x;
  const long N1 = (long)B_ * C_ * NS_;
  if (t >= 2 * N1) return;
  int br = (int)(t / N1);
  long u = t - (long)br * N1;
  int n = (int)(u % NS_);
  int c = (int)((u / NS_) % C_);
  int b = (int)(u / ((long)NS_ * C_));
  int sid = sidxs[b * NS_ + n];
  const float* a = br ? ragg : dagg;
  float* g = br ? r0g : d0g;
  atomicAdd(&g[((size_t)b * C_ + c) * HWO_ + sid], a[((size_t)b * C_ + c) * NS_ + n]);
}

// ---- mixed grid NCHW fp32 -> NHWC bf16 ----
__global__ void k_cvt_mid(const float* __restrict__ d0g, const float* __restrict__ r0g,
                          __bf16* __restrict__ m0d, __bf16* __restrict__ m0r) {
  long t = (long)blockIdx.x * 256 + threadIdx.x;
  const long N1 = (long)B_ * HWO_ * C_;
  if (t >= 2 * N1) return;
  int br = (int)(t / N1);
  long u = t - (long)br * N1;
  int c = (int)(u % C_);
  long v = u / C_;
  int pix = (int)(v % HWO_);
  int b = (int)(v / HWO_);
  const float* src = br ? r0g : d0g;
  __bf16* dst = br ? m0r : m0d;
  dst[u] = f2bf(src[((size_t)b * C_ + c) * HWO_ + pix]);
}

// ---- stride-1 conv2 + residual + relu -> d_out, bf16 WMMA ----
__global__ __launch_bounds__(256, 1)
void k_conv_s1(const __bf16* __restrict__ m0d, const __bf16* __restrict__ m0r,
               const __bf16* __restrict__ wpk1,
               const float* __restrict__ dc2b, const float* __restrict__ rc2b,
               const float* __restrict__ d1g, const float* __restrict__ r1g,
               float* __restrict__ outAll) {
  const int br = blockIdx.y;
  const __bf16* x  = br ? m0r : m0d;
  const __bf16* ws = wpk1 + (size_t)br * 18 * 64 * 32;
  const float* bp  = br ? rc2b : dc2b;
  const float* res = br ? r1g : d1g;
  float* out = outAll + (size_t)br * (size_t)B_ * C_ * HWO_;

  __shared__ __align__(16) __bf16 wlds[18 * 64 * 32];    // 72 KB
  __shared__ __align__(16) __bf16 atile[2][128 * 64];    // 2 x 16 KB

  tdm_load_weights(ws, wlds, 18 * 64 * 32);

  const int pixBase = blockIdx.x * 128;
  const int lane = threadIdx.x & 31;
  const int wave = threadIdx.x >> 5;
  const int mrow = wave * 16 + (lane & 15);
  const int ako  = (lane & 16) ? 16 : 0;
  const int bko  = (lane & 16) ? 16 : 0;

  v8f zero = {0.f, 0.f, 0.f, 0.f, 0.f, 0.f, 0.f, 0.f};
  v8f acc[4];
  for (int j = 0; j < 4; ++j) acc[j] = zero;

  stage_s1(x, atile[0], pixBase, 0, 0);
  wait_async_lds();
  __syncthreads();

  for (int tap = 0; tap < 9; ++tap) {
    const int cur = tap & 1;
    if (tap + 1 < 9)
      stage_s1(x, atile[1 - cur], pixBase, (tap + 1) / 3, (tap + 1) % 3);

    for (int half = 0; half < 2; ++half) {
      const __bf16* arow = atile[cur] + mrow * 64 + half * 32;
      Frag af;
      af.u[0] = *(const uint4*)(arow + ako);
      af.u[1] = *(const uint4*)(arow + ako + 8);
      int chunk = tap * 2 + half;
      Frag bf[4];
      for (int nt = 0; nt < 4; ++nt) {
        int n = nt * 16 + (lane & 15);
        const __bf16* wb = wlds + ((chunk * 64 + n) * 32 + bko);
        bf[nt].u[0] = ((const uint4*)wb)[0];
        bf[nt].u[1] = ((const uint4*)wb)[1];
      }
      for (int nt = 0; nt < 4; ++nt)
        acc[nt] = __builtin_amdgcn_wmma_f32_16x16x32_bf16(
            false, af.v, false, bf[nt].v, (short)0, acc[nt], false, false);
    }

    wait_async_lds();
    __syncthreads();
  }

  for (int nt = 0; nt < 4; ++nt) {
    int ch = nt * 16 + (lane & 15);
    float bi = bp[ch];
    for (int j = 0; j < 8; ++j) {
      int m = j + ((lane & 16) ? 8 : 0);
      int p = pixBase + wave * 16 + m;
      int bb = p / HWO_, off = p - bb * HWO_;
      size_t idx = ((size_t)bb * C_ + ch) * HWO_ + off;
      out[idx] = fmaxf(acc[nt][j] + bi + res[idx], 0.0f);
    }
  }
}

extern "C" void kernel_launch(void* const* d_in, const int* in_sizes, int n_in,
                              void* d_out, int out_size, void* d_ws, size_t ws_size,
                              hipStream_t stream) {
  (void)in_sizes; (void)n_in; (void)out_size; (void)ws_size;
  const float* d_feat  = (const float*)d_in[0];
  const float* r_feat  = (const float*)d_in[1];
  const float* spoints = (const float*)d_in[2];
  const int*   masks   = (const int*)d_in[3];
  const float* dc0w = (const float*)d_in[4],  *dc0b = (const float*)d_in[5];
  const float* dc1w = (const float*)d_in[6],  *dc1b = (const float*)d_in[7];
  const float* dc2w = (const float*)d_in[8],  *dc2b = (const float*)d_in[9];
  const float* rc0w = (const float*)d_in[10], *rc0b = (const float*)d_in[11];
  const float* rc1w = (const float*)d_in[12], *rc1b = (const float*)d_in[13];
  const float* rc2w = (const float*)d_in[14], *rc2b = (const float*)d_in[15];
  const float* dw1 = (const float*)d_in[16], *db1 = (const float*)d_in[17];
  const float* dw2 = (const float*)d_in[18], *db2 = (const float*)d_in[19];
  const float* rw1 = (const float*)d_in[20], *rb1 = (const float*)d_in[21];
  const float* rw2 = (const float*)d_in[22], *rb2 = (const float*)d_in[23];
  const float* dbias = (const float*)d_in[24], *rbias = (const float*)d_in[25];
  const int* sidxs  = (const int*)d_in[26];
  const int* nnidxs = (const int*)d_in[27];

  char* w = (char*)d_ws;
  auto take = [&](size_t bytes) -> void* {
    void* p = (void*)w;
    w += (bytes + 255) & ~(size_t)255;
    return p;
  };
  __bf16* xd   = (__bf16*)take((size_t)B_ * H_ * W_ * CIN_ * 2);
  __bf16* xr   = (__bf16*)take((size_t)B_ * H_ * W_ * CIN_ * 2);
  __bf16* wpk2 = (__bf16*)take((size_t)4 * 9 * 64 * 32 * 2);
  __bf16* wpk1 = (__bf16*)take((size_t)2 * 18 * 64 * 32 * 2);
  float* d0g = (float*)take((size_t)B_ * C_ * HWO_ * 4);
  float* d1g = (float*)take((size_t)B_ * C_ * HWO_ * 4);
  float* r0g = (float*)take((size_t)B_ * C_ * HWO_ * 4);
  float* r1g = (float*)take((size_t)B_ * C_ * HWO_ * 4);
  float* dsg  = (float*)take((size_t)B_ * C_ * NS_ * 4);
  float* rsg  = (float*)take((size_t)B_ * C_ * NS_ * 4);
  float* dagg = (float*)take((size_t)B_ * C_ * NS_ * 4);
  float* ragg = (float*)take((size_t)B_ * C_ * NS_ * 4);
  __bf16* m0d = (__bf16*)take((size_t)B_ * HWO_ * C_ * 2);
  __bf16* m0r = (__bf16*)take((size_t)B_ * HWO_ * C_ * 2);

  k_pack<<<(2 * 73728 + 255) / 256, 256, 0, stream>>>(dc0w, dc1w, rc0w, rc1w, dc2w, rc2w, wpk2, wpk1);

  {
    long N = 2L * B_ * H_ * W_ * CIN_;
    k_cvt_in<<<(unsigned)((N + 255) / 256), 256, 0, stream>>>(d_feat, r_feat, xd, xr);
  }

  k_conv_s2<<<dim3(NPIX_ / 128, 2), 256, 0, stream>>>(xd, xr, wpk2, dc0b, dc1b, rc0b, rc1b,
                                                      d0g, d1g, r0g, r1g);

  k_gather<<<(2 * B_ * C_ * NS_) / 256, 256, 0, stream>>>(d0g, r0g, sidxs, dsg, rsg);

  k_attn<<<(B_ * NS_) / 8, 256, 0, stream>>>(dsg, rsg, spoints, nnidxs,
                                             dw1, db1, dw2, db2, rw1, rb1, rw2, rb2,
                                             dbias, rbias, dagg, ragg);

  {
    long N = 2L * B_ * C_ * HWO_;
    k_mix<<<(unsigned)((N + 255) / 256), 256, 0, stream>>>(masks, d0g, r0g);
  }

  k_scatter<<<(2 * B_ * C_ * NS_) / 256, 256, 0, stream>>>(sidxs, dagg, ragg, d0g, r0g);

  {
    long N = 2L * B_ * HWO_ * C_;
    k_cvt_mid<<<(unsigned)((N + 255) / 256), 256, 0, stream>>>(d0g, r0g, m0d, m0r);
  }

  k_conv_s1<<<dim3(NPIX_ / 128, 2), 256, 0, stream>>>(m0d, m0r, wpk1, dc2b, rc2b, d1g, r1g,
                                                      (float*)d_out);
}
